// MultiHeadAttention_39384850104909
// MI455X (gfx1250) — matrix operand order repaired and checked by reading, not compiled, not run on hardware
//
#include <hip/hip_runtime.h>
#include <hip/hip_bf16.h>

typedef __bf16 bf16;
typedef __attribute__((ext_vector_type(16))) __bf16 bf16x16;
typedef __attribute__((ext_vector_type(2)))  __bf16 bf16x2;
typedef __attribute__((ext_vector_type(8)))  float  f32x8;
typedef __attribute__((ext_vector_type(4)))  int    i32x4;

#define B_  2
#define T_  2048
#define D_  2048
#define H_  16
#define DK_ 128
#define N3D 6144

// ---------- gfx1250 async LDS path (guarded) ----------
#if __has_builtin(__builtin_amdgcn_global_load_async_to_lds_b128) && \
    __has_builtin(__builtin_amdgcn_s_wait_asynccnt)
#define HAVE_ASYNC_LDS 1
#endif

__device__ __forceinline__ void async_copy16(bf16* lds_dst, const bf16* gsrc) {
#if defined(HAVE_ASYNC_LDS)
  __builtin_amdgcn_global_load_async_to_lds_b128(
      (__attribute__((address_space(1))) i32x4*)gsrc,
      (__attribute__((address_space(3))) i32x4*)lds_dst, 0, 0);
#else
  *(uint4*)lds_dst = *(const uint4*)gsrc;
#endif
}
__device__ __forceinline__ void wait_async() {
#if defined(HAVE_ASYNC_LDS)
  __builtin_amdgcn_s_wait_asynccnt(0);
#endif
}

// packed f32->bf16 conversion for contiguous pairs (guarded)
__device__ __forceinline__ void cvt2_bf16(bf16* dst, float a, float b) {
#if __has_builtin(__builtin_amdgcn_cvt_pk_bf16_f32)
  *(bf16x2*)dst = __builtin_amdgcn_cvt_pk_bf16_f32(a, b);
#else
  dst[0] = (bf16)a; dst[1] = (bf16)b;
#endif
}

// ---------- WMMA helpers ----------
__device__ __forceinline__ f32x8 wmma_bf16(bf16x16 a, bf16x16 b, f32x8 c) {
  // D = A(16x32) * B(32x16) + C, f32 accumulate
  return __builtin_amdgcn_wmma_f32_16x16x32_bf16(false, a, false, b, (short)0, c,
                                                 false, false);
}

union FragU {
  bf16x16 v;
  uint4   q[2];
};
__device__ __forceinline__ bf16x16 load_frag2(const bf16* p0, const bf16* p1) {
  FragU f;
  f.q[0] = *(const uint4*)p0;
  f.q[1] = *(const uint4*)p1;
  return f.v;
}
// A-fragment (16x32): k runs [hh*8, +8) and [16+hh*8, +8)   (ISA 7.12.2)
__device__ __forceinline__ bf16x16 load_frag_a(const bf16* row, int hh) {
  return load_frag2(row + hh * 8, row + 16 + hh * 8);
}
// B-fragment (32x16): k runs [hh*8, +8) and [16+hh*8, +8), same as A
__device__ __forceinline__ bf16x16 load_frag_b(const bf16* col, int hh) {
  return load_frag2(col + hh * 8, col + 16 + hh * 8);
}

// ---------- GEMM tiling ----------
#define BM 128
#define BN 128
#define BK 32
#define LDA  (BK + 8)   // row-major A tile stride, 80B rows (16B aligned)
#define LDBT (BK + 8)   // column-major B tile stride

// =====================================================================
// Kernel 1: qkv = x @ Wqkv -> Q/K/V [B,H,T,DK] bf16 (q pre-scaled DK^-0.5)
// =====================================================================
__global__ __launch_bounds__(256)
void qkv_gemm_kernel(const float* __restrict__ x, const float* __restrict__ Wqkv,
                     bf16* __restrict__ Qws, bf16* __restrict__ Kws,
                     bf16* __restrict__ Vws)
{
  __shared__ __align__(16) bf16 As [BM * LDA];    // 128 x (32+8) row-major
  __shared__ __align__(16) bf16 Bst[BN * LDBT];   // 128 x (32+8) col-major (n,k)

  const int tid  = threadIdx.x;
  const int lane = tid & 31;
  const int wid  = tid >> 5;
  const int hh   = lane >> 4;
  const int ln   = lane & 15;

  const int n0 = blockIdx.x * BN;
  const int m0 = blockIdx.y * BM;
  const int wm = wid >> 1;
  const int wn = wid & 1;

  const f32x8 zero8 = {0.f,0.f,0.f,0.f,0.f,0.f,0.f,0.f};
  f32x8 acc[2][4];
#pragma unroll
  for (int i = 0; i < 2; ++i)
#pragma unroll
    for (int j = 0; j < 4; ++j) acc[i][j] = zero8;

  for (int k0 = 0; k0 < D_; k0 += BK) {
    // stage A row-major: 128x32 f32 -> bf16
    {
      int r = tid >> 3, c = (tid & 7) * 4;
#pragma unroll
      for (int p = 0; p < 4; ++p) {
        int row = r + p * 32;
        const float4 v = *(const float4*)(&x[(size_t)(m0 + row) * D_ + k0 + c]);
        bf16* dst = &As[row * LDA + c];
        cvt2_bf16(dst + 0, v.x, v.y);
        cvt2_bf16(dst + 2, v.z, v.w);
      }
      // stage B column-major: Bst[n][k]
      int r2 = tid >> 5, c2 = (tid & 31) * 4;
#pragma unroll
      for (int p = 0; p < 4; ++p) {
        int row = r2 + p * 8;   // k index
        const float4 v = *(const float4*)(&Wqkv[(size_t)(k0 + row) * N3D + n0 + c2]);
        Bst[(c2 + 0) * LDBT + row] = (bf16)v.x;
        Bst[(c2 + 1) * LDBT + row] = (bf16)v.y;
        Bst[(c2 + 2) * LDBT + row] = (bf16)v.z;
        Bst[(c2 + 3) * LDBT + row] = (bf16)v.w;
      }
      if (k0 + BK < D_) {   // prefetch next tiles into cache
        __builtin_prefetch(&x[(size_t)(m0 + r) * D_ + k0 + BK + c], 0, 1);
        __builtin_prefetch(&Wqkv[(size_t)(k0 + BK + r2) * N3D + n0 + c2], 0, 1);
      }
    }
    __syncthreads();

    bf16x16 afrag[2];
#pragma unroll
    for (int im = 0; im < 2; ++im)
      afrag[im] = load_frag_a(&As[(wm * 32 + im * 16 + ln) * LDA], hh);
    bf16x16 bfrag[4];
#pragma unroll
    for (int in = 0; in < 4; ++in)
      bfrag[in] = load_frag_b(&Bst[(wn * 64 + in * 16 + ln) * LDBT], hh);
#pragma unroll
    for (int im = 0; im < 2; ++im)
#pragma unroll
      for (int in = 0; in < 4; ++in)
        acc[im][in] = wmma_bf16(afrag[im], bfrag[in], acc[im][in]);

    __syncthreads();
  }

  // Epilogue: scatter into Q/K/V head-major layout.  ntb is wave-uniform.
  const float qscale = 0.08838834764831845f;  // DK^-0.5
#pragma unroll
  for (int im = 0; im < 2; ++im) {
#pragma unroll
    for (int in = 0; in < 4; ++in) {
      const int ntb   = n0 + wn * 64 + in * 16;
      const int t128  = __builtin_amdgcn_readfirstlane(ntb >> 7); // 0..47
      const int h     = (t128 * 86) >> 8;       // /3, exact for 0..127
      const int which = t128 - 3 * h;           // 0=q 1=k 2=v
      const int d     = (ntb & 127) + ln;
      bf16* dstbase = (which == 0) ? Qws : (which == 1) ? Kws : Vws;
      const float scale = (which == 0) ? qscale : 1.0f;
#pragma unroll
      for (int j = 0; j < 8; ++j) {
        const int m = m0 + wm * 32 + im * 16 + j + hh * 8;
        const int b = m >> 11;            // / T_
        const int t = m & (T_ - 1);
        dstbase[(((size_t)b * H_ + h) * T_ + t) * DK_ + d] =
            (bf16)(acc[im][in][j] * scale);
      }
    }
  }
}

// =====================================================================
// Kernel 2: in-place RoPE on Q and K
// =====================================================================
__global__ __launch_bounds__(256)
void rope_kernel(bf16* __restrict__ Qws, bf16* __restrict__ Kws,
                 const float* __restrict__ freqs)
{
  const int PAIRS = DK_ / 2;  // 64
  int idx = blockIdx.x * blockDim.x + threadIdx.x;
  int pr = idx & (PAIRS - 1);
  int t  = (idx >> 6)  & (T_ - 1);
  int h  = (idx >> 17) & (H_ - 1);
  int b  = idx >> 21;

  float ang = freqs[((size_t)b * T_ + t) * PAIRS + pr];
  float s, c;
  sincosf(ang, &s, &c);

  size_t base = (((size_t)b * H_ + h) * T_ + t) * DK_ + 2 * pr;
  {
    float xr = (float)Qws[base], xi = (float)Qws[base + 1];
    cvt2_bf16(&Qws[base], xr * c - xi * s, xr * s + xi * c);
  }
  {
    float xr = (float)Kws[base], xi = (float)Kws[base + 1];
    cvt2_bf16(&Kws[base], xr * c - xi * s, xr * s + xi * c);
  }
}

// =====================================================================
// Kernel 3: causal flash attention per (b,h).
// K tile: async global->LDS copy (row-major).  V tile: transposed staging.
// =====================================================================
#define FA_BM 128
#define FA_TK 32
#define LDK  (DK_ + 8)
#define LDVT (FA_TK + 8)
#define LDP  (FA_TK + 8)

__global__ __launch_bounds__(256)
void flash_attn_kernel(const bf16* __restrict__ Qws, const bf16* __restrict__ Kws,
                       const bf16* __restrict__ Vws, bf16* __restrict__ Ows)
{
  __shared__ __align__(16) bf16 Ks [FA_TK * LDK];
  __shared__ __align__(16) bf16 Vst[DK_ * LDVT];
  __shared__ __align__(16) bf16 Ps [8 * 16 * LDP];

  const int tid  = threadIdx.x;
  const int lane = tid & 31;
  const int wid  = tid >> 5;
  const int hh   = lane >> 4;
  const int ln   = lane & 15;

  const int bh    = blockIdx.y;
  const int q0blk = blockIdx.x * FA_BM;
  const int q0w   = q0blk + wid * 16;

  const bf16* Qbh = Qws + (size_t)bh * T_ * DK_;
  const bf16* Kbh = Kws + (size_t)bh * T_ * DK_;
  const bf16* Vbh = Vws + (size_t)bh * T_ * DK_;

  bf16x16 qfrag[4];
  {
    const bf16* qrow = &Qbh[(size_t)(q0w + ln) * DK_];
#pragma unroll
    for (int kk = 0; kk < 4; ++kk)
      qfrag[kk] = load_frag_a(qrow + kk * 32, hh);
  }

  const f32x8 zero8 = {0.f,0.f,0.f,0.f,0.f,0.f,0.f,0.f};
  f32x8 Oacc[8];
  float mrow[8], lrow[8];
#pragma unroll
  for (int i = 0; i < 8; ++i) { Oacc[i] = zero8; mrow[i] = -1e30f; lrow[i] = 0.f; }

  const int nkt = (q0blk + FA_BM) / FA_TK;  // causal: keys 0 .. q0blk+127
  for (int kt = 0; kt < nkt; ++kt) {
    const int k0 = kt * FA_TK;
    {
      int r = tid >> 3;               // 0..31 (key row)
      int c = (tid & 7) * 16;         // 32B per thread
      // K: straight bf16 copy -> async load to LDS (ASYNCcnt path)
      const bf16* kg = &Kbh[(size_t)(k0 + r) * DK_ + c];
      async_copy16(&Ks[r * LDK + c],     kg);
      async_copy16(&Ks[r * LDK + c + 8], kg + 8);
      // V: needs transpose -> VGPR round trip + b16 scatter
      union { uint4 q[2]; bf16 h[16]; } vv;
      const uint4* vs = (const uint4*)&Vbh[(size_t)(k0 + r) * DK_ + c];
      vv.q[0] = vs[0]; vv.q[1] = vs[1];
#pragma unroll
      for (int i = 0; i < 16; ++i)
        Vst[(c + i) * LDVT + r] = vv.h[i];   // Vst[d][k]
      if (kt + 1 < nkt) {
        __builtin_prefetch(&Kbh[(size_t)(k0 + FA_TK + r) * DK_ + c], 0, 1);
        __builtin_prefetch(&Vbh[(size_t)(k0 + FA_TK + r) * DK_ + c], 0, 1);
      }
    }
    wait_async();
    __syncthreads();

    // S = Q K^T : 16x32 as two 16x16 tiles
    f32x8 s0 = zero8, s1 = zero8;
#pragma unroll
    for (int kk = 0; kk < 4; ++kk) {
      bf16x16 b0 = load_frag_b(&Ks[(ln)      * LDK + kk * 32], hh);
      bf16x16 b1 = load_frag_b(&Ks[(16 + ln) * LDK + kk * 32], hh);
      s0 = wmma_bf16(qfrag[kk], b0, s0);
      s1 = wmma_bf16(qfrag[kk], b1, s1);
    }

    // causal mask + online softmax (row reductions within 16-lane halves)
    float pv[16];
#pragma unroll
    for (int j = 0; j < 8; ++j) {
      const int qi = q0w + j + hh * 8;
      float a = s0[j]; if (k0 + ln > qi)      a = -1e30f;
      float b = s1[j]; if (k0 + 16 + ln > qi) b = -1e30f;
      float mt = fmaxf(a, b);
#pragma unroll
      for (int off = 1; off < 16; off <<= 1)
        mt = fmaxf(mt, __shfl_xor(mt, off, 32));
      const float mnew  = fmaxf(mrow[j], mt);
      const float alpha = __expf(mrow[j] - mnew);
      const float p0 = __expf(a - mnew);
      const float p1 = __expf(b - mnew);
      float rs = p0 + p1;
#pragma unroll
      for (int off = 1; off < 16; off <<= 1)
        rs += __shfl_xor(rs, off, 32);
      lrow[j] = lrow[j] * alpha + rs;
      mrow[j] = mnew;
#pragma unroll
      for (int dt = 0; dt < 8; ++dt) Oacc[dt][j] *= alpha;
      pv[j] = p0; pv[j + 8] = p1;
    }

    // P: C-layout -> A-layout via per-wave LDS scratch
    bf16* Pw = &Ps[wid * 16 * LDP];
#pragma unroll
    for (int j = 0; j < 8; ++j) {
      const int m = j + hh * 8;
      Pw[m * LDP + ln]      = (bf16)pv[j];
      Pw[m * LDP + 16 + ln] = (bf16)pv[j + 8];
    }
    bf16x16 pfrag = load_frag_a(&Ps[wid * 16 * LDP + ln * LDP], hh);

    // O += P * V
#pragma unroll
    for (int dt = 0; dt < 8; ++dt) {
      bf16x16 vfrag = load_frag_b(&Vst[(dt * 16 + ln) * LDVT], hh);
      Oacc[dt] = wmma_bf16(pfrag, vfrag, Oacc[dt]);
    }
    __syncthreads();
  }

  // normalize + store O bf16 at [b*T + q][h*DK + d]
  const int b = bh / H_, h = bh % H_;
#pragma unroll
  for (int j = 0; j < 8; ++j) {
    const int q = q0w + j + hh * 8;
    const float inv = 1.0f / lrow[j];
    const size_t rowoff = ((size_t)b * T_ + q) * D_ + h * DK_;
#pragma unroll
    for (int dt = 0; dt < 8; ++dt)
      Ows[rowoff + dt * 16 + ln] = (bf16)(Oacc[dt][j] * inv);
  }
}

// =====================================================================
// Kernel 4: out = O @ Wout   (f32 output)
// =====================================================================
__global__ __launch_bounds__(256)
void out_gemm_kernel(const bf16* __restrict__ Ows, const float* __restrict__ Wout,
                     float* __restrict__ out)
{
  __shared__ __align__(16) bf16 As [BM * LDA];
  __shared__ __align__(16) bf16 Bst[BN * LDBT];

  const int tid  = threadIdx.x;
  const int lane = tid & 31;
  const int wid  = tid >> 5;
  const int hh   = lane >> 4;
  const int ln   = lane & 15;

  const int n0 = blockIdx.x * BN;
  const int m0 = blockIdx.y * BM;
  const int wm = wid >> 1;
  const int wn = wid & 1;

  const f32x8 zero8 = {0.f,0.f,0.f,0.f,0.f,0.f,0.f,0.f};
  f32x8 acc[2][4];
#pragma unroll
  for (int i = 0; i < 2; ++i)
#pragma unroll
    for (int j = 0; j < 4; ++j) acc[i][j] = zero8;

  for (int k0 = 0; k0 < D_; k0 += BK) {
    {
      // stage A (bf16 already): async global->LDS copy
      int r = tid >> 1, c = (tid & 1) * 16;
      const bf16* ag = &Ows[(size_t)(m0 + r) * D_ + k0 + c];
      async_copy16(&As[r * LDA + c],     ag);
      async_copy16(&As[r * LDA + c + 8], ag + 8);
      // stage B column-major: Bst[n][k], 32x128 f32 -> bf16
      int r2 = tid >> 5, c2 = (tid & 31) * 4;
#pragma unroll
      for (int p = 0; p < 4; ++p) {
        int row = r2 + p * 8;   // k index
        const float4 v = *(const float4*)(&Wout[(size_t)(k0 + row) * D_ + n0 + c2]);
        Bst[(c2 + 0) * LDBT + row] = (bf16)v.x;
        Bst[(c2 + 1) * LDBT + row] = (bf16)v.y;
        Bst[(c2 + 2) * LDBT + row] = (bf16)v.z;
        Bst[(c2 + 3) * LDBT + row] = (bf16)v.w;
      }
      if (k0 + BK < D_) {
        __builtin_prefetch(&Ows[(size_t)(m0 + r) * D_ + k0 + BK + c], 0, 1);
        __builtin_prefetch(&Wout[(size_t)(k0 + BK + r2) * D_ + n0 + c2], 0, 1);
      }
    }
    wait_async();
    __syncthreads();

    bf16x16 afrag[2];
#pragma unroll
    for (int im = 0; im < 2; ++im)
      afrag[im] = load_frag_a(&As[(wm * 32 + im * 16 + ln) * LDA], hh);
    bf16x16 bfrag[4];
#pragma unroll
    for (int in = 0; in < 4; ++in)
      bfrag[in] = load_frag_b(&Bst[(wn * 64 + in * 16 + ln) * LDBT], hh);
#pragma unroll
    for (int im = 0; im < 2; ++im)
#pragma unroll
      for (int in = 0; in < 4; ++in)
        acc[im][in] = wmma_bf16(afrag[im], bfrag[in], acc[im][in]);

    __syncthreads();
  }

#pragma unroll
  for (int im = 0; im < 2; ++im)
#pragma unroll
    for (int in = 0; in < 4; ++in) {
      const int n = n0 + wn * 64 + in * 16 + ln;
#pragma unroll
      for (int j = 0; j < 8; ++j) {
        const int m = m0 + wm * 32 + im * 16 + j + hh * 8;
        out[(size_t)m * D_ + n] = acc[im][in][j];
      }
    }
}

// =====================================================================
extern "C" void kernel_launch(void* const* d_in, const int* in_sizes, int n_in,
                              void* d_out, int out_size, void* d_ws, size_t ws_size,
                              hipStream_t stream)
{
  (void)in_sizes; (void)n_in; (void)out_size; (void)ws_size;
  const float* x     = (const float*)d_in[0];
  const float* freqs = (const float*)d_in[1];
  const float* Wqkv  = (const float*)d_in[2];
  const float* Wout  = (const float*)d_in[3];
  float* out = (float*)d_out;

  char* ws = (char*)d_ws;
  const size_t elems = (size_t)B_ * H_ * T_ * DK_;      // 8,388,608
  bf16* Qws = (bf16*)(ws);
  bf16* Kws = (bf16*)(ws + 2 * elems);
  bf16* Vws = (bf16*)(ws + 4 * elems);
  bf16* Ows = (bf16*)(ws + 6 * elems);                  // 64MB total

  qkv_gemm_kernel<<<dim3(N3D / BN, (B_ * T_) / BM), 256, 0, stream>>>(
      x, Wqkv, Qws, Kws, Vws);
  rope_kernel<<<(B_ * H_ * T_ * (DK_ / 2)) / 256, 256, 0, stream>>>(
      Qws, Kws, freqs);
  flash_attn_kernel<<<dim3(T_ / FA_BM, B_ * H_), 256, 0, stream>>>(
      Qws, Kws, Vws, Ows);
  out_gemm_kernel<<<dim3(D_ / BN, (B_ * T_) / BM), 256, 0, stream>>>(
      Ows, Wout, out);
}
